// SelfAttentionMechanism_50792283242874
// MI455X (gfx1250) — matrix operand order repaired and checked by reading, not compiled, not run on hardware
//
#include <hip/hip_runtime.h>

typedef __attribute__((ext_vector_type(16))) __bf16 v16bf;
typedef __attribute__((ext_vector_type(8)))  __bf16 v8bf;
typedef __attribute__((ext_vector_type(8)))  float  v8f;

#define D_   1024
#define H_   16
#define DK_  64
#define S_   2048
#define B_   2
#define E_   8
#define MTOK 4096   // B_ * S_

// 0.125 (=1/sqrt(DK)) * log2(e): scores scaled into base-2 domain for exp2
#define SCALE_LOG2E 0.18033688011112042f

// ---------------------------------------------------------------------------
// helpers
// ---------------------------------------------------------------------------
static __device__ inline v16bf cat16(v8bf lo, v8bf hi) {
  struct P { v8bf a, b; } p{lo, hi};
  return __builtin_bit_cast(v16bf, p);
}

// A-fragment (16x32 bf16, MxK): lane L(0..15): M=L, elems0..15 <- K=ka..ka+15
// (contiguous); lanes 16..31 same M, K shifted by +16.
static __device__ inline v16bf load_a_frag(const __bf16* __restrict__ A,
                                           size_t row_off, int ka) {
  const v8bf* p0 = (const v8bf*)(A + row_off + ka);
  const v8bf* p1 = (const v8bf*)(A + row_off + ka + 8);
  return cat16(*p0, *p1);
}

static __device__ inline v16bf load_a_frag(const float* __restrict__ A,
                                           size_t row_off, int ka) {
  const float4* p0 = (const float4*)(A + row_off + ka);
  const float4* p1 = (const float4*)(A + row_off + ka + 8);
  float4 f0 = p0[0], f1 = p0[1], f2 = p1[0], f3 = p1[1];
  v16bf a;
  a[0]  = (__bf16)f0.x; a[1]  = (__bf16)f0.y; a[2]  = (__bf16)f0.z; a[3]  = (__bf16)f0.w;
  a[4]  = (__bf16)f1.x; a[5]  = (__bf16)f1.y; a[6]  = (__bf16)f1.z; a[7]  = (__bf16)f1.w;
  a[8]  = (__bf16)f2.x; a[9]  = (__bf16)f2.y; a[10] = (__bf16)f2.z; a[11] = (__bf16)f2.w;
  a[12] = (__bf16)f3.x; a[13] = (__bf16)f3.y; a[14] = (__bf16)f3.z; a[15] = (__bf16)f3.w;
  return a;
}

// ---------------------------------------------------------------------------
// weight prep: f32 -> bf16 conversion + expert averaging (mean is linear, so
// mean_e(x @ We[e].T + be[e]) == x @ mean(We).T + mean(be))
// ---------------------------------------------------------------------------
__global__ __launch_bounds__(256)
void prep_weights_kernel(const float* __restrict__ Wq, const float* __restrict__ Wk,
                         const float* __restrict__ Wv, const float* __restrict__ Wo,
                         const float* __restrict__ We, const float* __restrict__ be,
                         __bf16* __restrict__ wq, __bf16* __restrict__ wk,
                         __bf16* __restrict__ wv, __bf16* __restrict__ wo,
                         __bf16* __restrict__ wavg, float* __restrict__ bavg) {
  const int i = blockIdx.x * blockDim.x + threadIdx.x;
  const int DD = D_ * D_;
  if (i < DD) {
    wq[i] = (__bf16)Wq[i];
    wk[i] = (__bf16)Wk[i];
    wv[i] = (__bf16)Wv[i];
    wo[i] = (__bf16)Wo[i];
    float s = 0.f;
#pragma unroll
    for (int e = 0; e < E_; ++e) s += We[(size_t)e * DD + i];
    wavg[i] = (__bf16)(s * (1.0f / E_));
    if (i < D_) {
      float sb = 0.f;
#pragma unroll
      for (int e = 0; e < E_; ++e) sb += be[e * D_ + i];
      bavg[i] = sb * (1.0f / E_);
    }
  }
}

// ---------------------------------------------------------------------------
// WMMA GEMM:  C[M,N] = A[M,K] @ W[N,K]^T + bias[N]
// Both A and W are K-contiguous -> contiguous per-lane fragment loads.
// block = 256 threads = 8 waves; wave computes 32(M) x 64(N); block: 256x64.
// TRANS=true stores the result transposed (C[n,m]) -- used for V so the
// attention kernel can load V column-fragments as contiguous 32B vectors.
// ---------------------------------------------------------------------------
template <typename TIn, typename TOut, bool TRANS>
__global__ __launch_bounds__(256)
void gemm_bias_kernel(const TIn* __restrict__ A, const __bf16* __restrict__ W,
                      const float* __restrict__ bias, TOut* __restrict__ C,
                      int M, int N, int K) {
  const int lane   = threadIdx.x & 31;
  const int wv     = threadIdx.x >> 5;
  const int laneN  = lane & 15;
  const int laneHi = lane >> 4;

  const int m0 = blockIdx.y * 256 + wv * 32;
  const int n0 = blockIdx.x * 64;

  v8f acc[2][4] = {};
  const size_t arow0 = (size_t)(m0 + laneN) * K;
  const size_t arow1 = (size_t)(m0 + 16 + laneN) * K;

  for (int k0 = 0; k0 < K; k0 += 32) {
    // group all fragment loads, then all WMMAs (one wait, clause-able loads)
    const int ka = k0 + laneHi * 16;
    v16bf a0 = load_a_frag(A, arow0, ka);
    v16bf a1 = load_a_frag(A, arow1, ka);
    v16bf bw[4];
#pragma unroll
    for (int t = 0; t < 4; ++t) {
      // B-fragment (32x16, KxN): lane = N, elems = 16 contiguous K values
      const __bf16* wp = W + (size_t)(n0 + t * 16 + laneN) * K + k0 + laneHi * 16;
      bw[t] = *(const v16bf*)wp;
    }
    if (k0 + 32 < K) {
      __builtin_prefetch(A + arow0 + ka + 32, 0, 3);
      __builtin_prefetch(A + arow1 + ka + 32, 0, 3);
      __builtin_prefetch(W + (size_t)(n0 + laneN) * K + k0 + 32, 0, 3);
    }
#pragma unroll
    for (int t = 0; t < 4; ++t)
      acc[0][t] = __builtin_amdgcn_wmma_f32_16x16x32_bf16(
          false, a0, false, bw[t], (short)0, acc[0][t], false, false);
#pragma unroll
    for (int t = 0; t < 4; ++t)
      acc[1][t] = __builtin_amdgcn_wmma_f32_16x16x32_bf16(
          false, a1, false, bw[t], (short)0, acc[1][t], false, false);
  }

#pragma unroll
  for (int r2 = 0; r2 < 2; ++r2) {
#pragma unroll
    for (int t = 0; t < 4; ++t) {
      const int n = n0 + t * 16 + laneN;
      const float bn = bias[n];
#pragma unroll
      for (int e = 0; e < 8; ++e) {
        const int m = m0 + r2 * 16 + e + 8 * laneHi;  // lanes16-31 hold M+8
        const float v = acc[r2][t][e] + bn;
        const size_t idx = TRANS ? ((size_t)n * M + m) : ((size_t)m * N + n);
        C[idx] = (TOut)v;
      }
    }
  }
}

// ---------------------------------------------------------------------------
// Flash attention: 1 wave per 16-query tile of one (b,h); online softmax in
// base-2 domain. 4 fully independent waves per block; per-wave LDS slices +
// in-order DS within a wave => no block barriers needed.
// V is consumed TRANSPOSED (Vt[d][token]) so B-fragments are contiguous.
// ---------------------------------------------------------------------------
__global__ __launch_bounds__(128)
void attention_kernel(const __bf16* __restrict__ Qb, const __bf16* __restrict__ Kb,
                      const __bf16* __restrict__ Vt, __bf16* __restrict__ AO) {
  __shared__ __align__(16) float  sS[4][16][33];   // score chunk, padded
  __shared__ __align__(16) __bf16 sP[4][16][32];   // exp2(scores) in bf16
  __shared__ __align__(16) float  sFac[4][16];     // per-row rescale factor

  const int lane   = threadIdx.x & 31;
  const int wv     = threadIdx.x >> 5;
  const int laneN  = lane & 15;
  const int laneHi = lane >> 4;

  const int wid = blockIdx.x * 4 + wv;      // 0..4095
  const int qt  = wid & (S_ / 16 - 1);      // query tile
  const int hb  = wid >> 7;
  const int h   = hb & (H_ - 1);
  const int b   = hb >> 4;

  const int rowbase = b * S_ + qt * 16;     // global token row of query 0
  const int colbase = h * DK_;              // feature column base for this head

  // preload Q fragments (DK=64 -> two 16x32 A-fragments), unscaled;
  // 1/sqrt(DK)*log2(e) is folded into the softmax (base-2 exponentials).
  v16bf qa[2];
#pragma unroll
  for (int kc = 0; kc < 2; ++kc) {
    const size_t off = (size_t)(rowbase + laneN) * D_ + colbase;
    qa[kc] = load_a_frag(Qb, off, kc * 32 + laneHi * 16);
  }

  v8f o[4] = {};                  // output accum: 16 queries x 64 dims
  float r_m = -__builtin_inff();  // running max (valid in lanes 0..15)
  float r_l = 0.f;                // running sum  (valid in lanes 0..15)

  for (int kt = 0; kt < S_ / 32; ++kt) {
    const int k0g = kt * 32;  // global key base for this chunk

    // ---- load all K fragments for this chunk, then do all 4 score WMMAs
    v16bf kf[2][2];
#pragma unroll
    for (int j = 0; j < 2; ++j)
#pragma unroll
      for (int kc = 0; kc < 2; ++kc)
        kf[j][kc] = *(const v16bf*)(Kb +
            (size_t)(b * S_ + k0g + j * 16 + laneN) * D_ +
            colbase + kc * 32 + laneHi * 16);

    if (kt + 1 < S_ / 32) {   // prefetch next chunk (K rows + Vt rows)
      __builtin_prefetch(Kb + (size_t)(b * S_ + k0g + 32 + laneN) * D_ + colbase, 0, 3);
      __builtin_prefetch(Vt + (size_t)(colbase + laneN) * MTOK + b * S_ + k0g + 32, 0, 3);
    }

    v8f c[2];
#pragma unroll
    for (int j = 0; j < 2; ++j) {
      v8f cj = {};
      cj = __builtin_amdgcn_wmma_f32_16x16x32_bf16(false, qa[0], false, kf[j][0],
                                                   (short)0, cj, false, false);
      cj = __builtin_amdgcn_wmma_f32_16x16x32_bf16(false, qa[1], false, kf[j][1],
                                                   (short)0, cj, false, false);
      c[j] = cj;
    }

    // spill C-fragments to LDS as [row][key] f32 (in-order DS, same wave)
#pragma unroll
    for (int j = 0; j < 2; ++j)
#pragma unroll
      for (int e = 0; e < 8; ++e)
        sS[wv][e + 8 * laneHi][laneN + 16 * j] = c[j][e];

    // ---- online softmax update (lanes 0..15 each own one query row)
    if (lane < 16) {
      const int r = lane;
      float sv[32], mx = -__builtin_inff();
#pragma unroll
      for (int j = 0; j < 32; ++j) {
        sv[j] = sS[wv][r][j] * SCALE_LOG2E;   // base-2 logits
        mx = fmaxf(mx, sv[j]);
      }
      const float mnew = fmaxf(r_m, mx);
      const float fac  = __builtin_exp2f(r_m - mnew);
      float sum = 0.f;
#pragma unroll
      for (int c4 = 0; c4 < 4; ++c4) {
        v8bf pk;
#pragma unroll
        for (int j = 0; j < 8; ++j) {
          const float p = __builtin_exp2f(sv[c4 * 8 + j] - mnew);
          sum += p;
          pk[j] = (__bf16)p;
        }
        *(v8bf*)&sP[wv][r][c4 * 8] = pk;
      }
      r_l = r_l * fac + sum;
      r_m = mnew;
      sFac[wv][r] = fac;
    }

    // ---- rescale accumulators by per-row factor
#pragma unroll
    for (int t = 0; t < 4; ++t) {
#pragma unroll
      for (int e = 0; e < 8; ++e) o[t][e] *= sFac[wv][e + 8 * laneHi];
    }

    // ---- P as A-fragment (16x32) from LDS
    v16bf pa = cat16(*(const v8bf*)&sP[wv][laneN][laneHi * 16],
                     *(const v8bf*)&sP[wv][laneN][laneHi * 16 + 8]);

    // ---- O += P @ V_chunk; Vt gives contiguous 32B B-fragments
    v16bf vf[4];
#pragma unroll
    for (int t = 0; t < 4; ++t)
      vf[t] = *(const v16bf*)(Vt +
          (size_t)(colbase + t * 16 + laneN) * MTOK +
          b * S_ + k0g + laneHi * 16);
#pragma unroll
    for (int t = 0; t < 4; ++t)
      o[t] = __builtin_amdgcn_wmma_f32_16x16x32_bf16(
          false, pa, false, vf[t], (short)0, o[t], false, false);
  }

  // ---- finalize: divide by softmax denominator, write bf16
  if (lane < 16) sFac[wv][lane] = 1.0f / r_l;
#pragma unroll
  for (int t = 0; t < 4; ++t) {
#pragma unroll
    for (int e = 0; e < 8; ++e) {
      const int r = e + 8 * laneHi;
      const float v = o[t][e] * sFac[wv][r];
      AO[(size_t)(rowbase + r) * D_ + colbase + t * 16 + laneN] = (__bf16)v;
    }
  }
}

// ---------------------------------------------------------------------------
// launch
// ---------------------------------------------------------------------------
extern "C" void kernel_launch(void* const* d_in, const int* in_sizes, int n_in,
                              void* d_out, int out_size, void* d_ws, size_t ws_size,
                              hipStream_t stream) {
  const float* query = (const float*)d_in[0];
  const float* key   = (const float*)d_in[1];
  const float* value = (const float*)d_in[2];
  const float* Wq = (const float*)d_in[3];  const float* bq = (const float*)d_in[4];
  const float* Wk = (const float*)d_in[5];  const float* bk = (const float*)d_in[6];
  const float* Wv = (const float*)d_in[7];  const float* bv = (const float*)d_in[8];
  const float* We = (const float*)d_in[9];  const float* be = (const float*)d_in[10];
  const float* Wo = (const float*)d_in[11]; const float* bo = (const float*)d_in[12];
  float* out = (float*)d_out;

  // workspace carve-up (bf16 weights + bf16 activations), ~50 MB total
  const size_t DD = (size_t)D_ * D_;
  const size_t T  = (size_t)MTOK * D_;
  char* ws = (char*)d_ws;
  __bf16* wq   = (__bf16*)ws;
  __bf16* wk   = wq + DD;
  __bf16* wv   = wk + DD;
  __bf16* wo   = wv + DD;
  __bf16* wavg = wo + DD;
  float*  bavg = (float*)(wavg + DD);
  __bf16* Qb   = (__bf16*)((char*)bavg + 4096);
  __bf16* Kb   = Qb + T;
  __bf16* Vt   = Kb + T;   // V projection stored TRANSPOSED: [D_][MTOK]
  __bf16* AO   = Vt + T;
  __bf16* EX   = AO + T;

  prep_weights_kernel<<<(D_ * D_) / 256, 256, 0, stream>>>(
      Wq, Wk, Wv, Wo, We, be, wq, wk, wv, wo, wavg, bavg);

  dim3 gg(D_ / 64, MTOK / 256);  // (N tiles, M tiles)
  gemm_bias_kernel<float, __bf16, false><<<gg, 256, 0, stream>>>(query, wq, bq, Qb, MTOK, D_, D_);
  gemm_bias_kernel<float, __bf16, false><<<gg, 256, 0, stream>>>(key,   wk, bk, Kb, MTOK, D_, D_);
  gemm_bias_kernel<float, __bf16, true ><<<gg, 256, 0, stream>>>(value, wv, bv, Vt, MTOK, D_, D_);

  attention_kernel<<<(B_ * H_ * (S_ / 16)) / 4, 128, 0, stream>>>(Qb, Kb, Vt, AO);

  gemm_bias_kernel<__bf16, __bf16, false><<<gg, 256, 0, stream>>>(AO, wavg, bavg, EX, MTOK, D_, D_);
  gemm_bias_kernel<__bf16, float , false><<<gg, 256, 0, stream>>>(EX, wo,   bo,   out, MTOK, D_, D_);
}
